// KalmanGainCalculatorMambaBlock_14164802142693
// MI455X (gfx1250) — compile-verified, matrix operand order checked by reading, not run on hardware
//
#include <hip/hip_runtime.h>
#include <math.h>

// ---------------- problem constants (from reference) ----------------
#define CC   96      // d_model
#define DIx  192     // d_inner
#define NS   16      // d_state
#define RR   6       // dt_rank
#define KK   4       // scan directions
#define BB   2
#define HH   48
#define WW   48
#define LL   (HH*WW)         // 2304
#define ROWS (BB*LL)         // 4608
#define CDBL (RR + 2*NS)     // 38

typedef __attribute__((ext_vector_type(16))) _Float16 v16h;
typedef __attribute__((ext_vector_type(8)))  float    v8f;

__device__ __forceinline__ int lane_id() { return threadIdx.x & 31; }

__device__ __forceinline__ float wave_sum(float v) {
#pragma unroll
  for (int m = 16; m >= 1; m >>= 1) v += __shfl_xor(v, m, 32);
  return v;
}

// ---- WMMA fragment loaders (CDNA5 16x16x32 f16 layouts, ISA 7.12.2) ----
// A tile: 16(M) x 32(K), row-major f16 source with leading dim lda.
__device__ __forceinline__ v16h frag_a_rm(const _Float16* A, int lda) {
  const int l = lane_id(), m = l & 15, hf = l >> 4;
  v16h a;
#pragma unroll
  for (int e = 0; e < 16; ++e) {
    const int k = (e & 7) + 8*hf + 16*(e >> 3);
    a[e] = A[m * lda + k];
  }
  return a;
}
// B tile: B[k][n] = W[n][k] from f32 row-major W (N x Kd), leading dim ldw.
__device__ __forceinline__ v16h frag_b_wt(const float* W, int ldw) {
  const int l = lane_id(), n = l & 15, hf = l >> 4;
  v16h b;
#pragma unroll
  for (int e = 0; e < 16; ++e) {
    const int k = (e & 7) + 8*hf + 16*(e >> 3);
    b[e] = (_Float16)W[n * ldw + k];
  }
  return b;
}

// scan-direction index permutation: position in flattened L for direction k
__device__ __forceinline__ int scan_perm(int k, int l) {
  switch (k & 3) {
    case 0:  return l;
    case 1:  return (l % WW) * WW + (l / WW);         // H == W == 48
    case 2:  return LL - 1 - l;
    default: { int lr = LL - 1 - l; return (lr % WW) * WW + (lr / WW); }
  }
}

// ---------------- K1: transpose (B,C,H,W)->(B,L,C), LayerNorm1 ----------------
__global__ void k_ln1(const float* __restrict__ x, const float* __restrict__ w,
                      const float* __restrict__ bias, float* __restrict__ tx,
                      _Float16* __restrict__ h1) {
  int row = blockIdx.x * 4 + (threadIdx.x >> 5);
  if (row >= ROWS) return;
  int b = row / LL, l = row % LL;
  int lane = lane_id();
  float v[3], s = 0.f, s2 = 0.f;
#pragma unroll
  for (int j = 0; j < 3; ++j) {
    int c = lane + 32*j;
    float t = x[((size_t)(b*CC + c))*LL + l];
    v[j] = t; s += t; s2 += t*t;
  }
  s = wave_sum(s); s2 = wave_sum(s2);
  float mu = s * (1.0f/CC);
  float var = s2 * (1.0f/CC) - mu*mu;
  float rstd = rsqrtf(var + 1e-5f);
#pragma unroll
  for (int j = 0; j < 3; ++j) {
    int c = lane + 32*j;
    tx[(size_t)row*CC + c] = v[j];
    h1[(size_t)row*CC + c] = (_Float16)((v[j]-mu)*rstd*w[c] + bias[c]);
  }
}

// ---------------- K2: in_proj GEMM  [4608x96]x[96x384] -> xz ----------------
__global__ void k_gemm_inproj(const _Float16* __restrict__ h1,
                              const float* __restrict__ w,     // (384,96)
                              float* __restrict__ xz) {        // (ROWS, 384)
  const int NT = (2*DIx)/16;                                   // 24
  int tile = blockIdx.x * 4 + (threadIdx.x >> 5);
  if (tile >= (ROWS/16)*NT) return;
  int mt = tile / NT, nt = tile % NT;
  v8f acc = {};
#pragma unroll
  for (int ks = 0; ks < CC; ks += 32) {
    v16h a = frag_a_rm(h1 + (size_t)mt*16*CC + ks, CC);
    v16h b = frag_b_wt(w + (size_t)nt*16*CC + ks, CC);
    acc = __builtin_amdgcn_wmma_f32_16x16x32_f16(false, a, false, b, (short)0, acc, false, false);
  }
  int lane = lane_id(), n = nt*16 + (lane & 15), hf = lane >> 4;
#pragma unroll
  for (int r = 0; r < 8; ++r) {
    int m = mt*16 + r + 8*hf;
    xz[(size_t)m*(2*DIx) + n] = acc[r];
  }
}

// ---------------- K3: depthwise 3x3 conv + bias + SiLU -> xch f16 (B,DI,L) ----------------
__global__ void k_conv(const float* __restrict__ xz, const float* __restrict__ cw,
                       const float* __restrict__ cb, _Float16* __restrict__ xch) {
  int t = blockIdx.x * blockDim.x + threadIdx.x;
  if (t >= BB*DIx*LL) return;
  int l = t % LL; int rest = t / LL; int d = rest % DIx; int b = rest / DIx;
  int h = l / WW, w = l % WW;
  float acc = cb[d];
#pragma unroll
  for (int i = 0; i < 3; ++i)
#pragma unroll
    for (int j = 0; j < 3; ++j) {
      int hh = h + i - 1, ww = w + j - 1;
      if (hh >= 0 && hh < HH && ww >= 0 && ww < WW)
        acc += xz[((size_t)b*LL + hh*WW + ww)*(2*DIx) + d] * cw[d*9 + i*3 + j];
    }
  acc = acc / (1.0f + __expf(-acc));          // SiLU
  xch[((size_t)b*DIx + d)*LL + l] = (_Float16)acc;
}

// ---------------- K4: x_proj GEMM (per b,k; A gathers through scan permutation) ----------------
__global__ void k_gemm_xproj(const _Float16* __restrict__ xch,  // (B,DI,L) f16
                             const float* __restrict__ xw,      // (K,38,DI)
                             float* __restrict__ xdbl) {        // (B,K,L,38)
  const int MT = LL/16, NT = 3;                                 // 144, 48-padded cols
  int tile = blockIdx.x * 4 + (threadIdx.x >> 5);
  if (tile >= BB*KK*MT*NT) return;
  int nt = tile % NT; tile /= NT;
  int mt = tile % MT; tile /= MT;
  int k  = tile % KK; int b = tile / KK;
  const int lane = lane_id(), m = lane & 15, hf = lane >> 4;
  const int lrow = mt*16 + m;
  const int pl = scan_perm(k, lrow);
  const _Float16* abase = xch + (size_t)b*DIx*LL + pl;
  const int n = nt*16 + m;
  v8f acc = {};
#pragma unroll
  for (int ks = 0; ks < DIx; ks += 32) {
    v16h a, bf;
#pragma unroll
    for (int e = 0; e < 16; ++e) {
      int kk = ks + (e & 7) + 8*hf + 16*(e >> 3);
      a[e]  = abase[(size_t)kk * LL];
      bf[e] = (n < CDBL) ? (_Float16)xw[((size_t)(k*CDBL + n))*DIx + kk] : (_Float16)0.0f;
    }
    acc = __builtin_amdgcn_wmma_f32_16x16x32_f16(false, a, false, bf, (short)0, acc, false, false);
  }
  if (n < CDBL) {
#pragma unroll
    for (int r = 0; r < 8; ++r) {
      int lr = mt*16 + r + 8*hf;
      xdbl[((size_t)(b*KK + k)*LL + lr)*CDBL + n] = acc[r];
    }
  }
}

// ---------------- K5: dt_proj GEMM (K=6 zero-padded to 32) + bias + softplus ----------------
__global__ void k_gemm_dtproj(const float* __restrict__ xdbl,   // (B,K,L,38)
                              const float* __restrict__ dtw,    // (K,DI,R)
                              const float* __restrict__ dtb,    // (K,DI)
                              float* __restrict__ delta) {      // (B,K,DI,L)
  const int MT = LL/16, NT = DIx/16;                            // 144, 12
  int tile = blockIdx.x * 4 + (threadIdx.x >> 5);
  if (tile >= BB*KK*MT*NT) return;
  int nt = tile % NT; tile /= NT;
  int mt = tile % MT; tile /= MT;
  int k = tile % KK, b = tile / KK;
  const int lane = lane_id(), m = lane & 15, hf = lane >> 4;
  const int lrow = mt*16 + m;
  const float* arow = xdbl + ((size_t)(b*KK + k)*LL + lrow)*CDBL;
  const int n = nt*16 + m;                                      // d index
  v16h a, bf;
#pragma unroll
  for (int e = 0; e < 16; ++e) {
    int kk = (e & 7) + 8*hf + 16*(e >> 3);
    a[e]  = (kk < RR) ? (_Float16)arow[kk] : (_Float16)0.0f;
    bf[e] = (kk < RR) ? (_Float16)dtw[((size_t)(k*DIx + n))*RR + kk] : (_Float16)0.0f;
  }
  v8f acc = {};
  acc = __builtin_amdgcn_wmma_f32_16x16x32_f16(false, a, false, bf, (short)0, acc, false, false);
  float bias = dtb[k*DIx + n];
#pragma unroll
  for (int r = 0; r < 8; ++r) {
    int lr = mt*16 + r + 8*hf;
    float v = acc[r] + bias;
    v = (v > 20.0f) ? v : log1pf(__expf(v));                    // softplus
    delta[((size_t)(b*KK + k)*DIx + n)*LL + lr] = v;
  }
}

// ---------------- K6: selective scan (state h[16] in VGPRs, sequential over L) ----------------
__global__ void k_scan(const float* __restrict__ delta,   // (B,K,DI,L)
                       const _Float16* __restrict__ xch,  // (B,DI,L) — u via perm
                       const float* __restrict__ xdbl,    // (B,K,L,38) — Bm,Cm
                       const float* __restrict__ A_log,   // (K,DI,N)
                       const float* __restrict__ Ds,      // (K,DI)
                       float* __restrict__ yb) {          // (B,K,DI,L)
  int t = blockIdx.x * blockDim.x + threadIdx.x;
  if (t >= BB*KK*DIx) return;
  int d = t % DIx; int rest = t / DIx; int k = rest % KK; int b = rest / KK;
  float a[NS], h[NS];
#pragma unroll
  for (int n = 0; n < NS; ++n) {
    a[n] = -__expf(A_log[((size_t)(k*DIx + d))*NS + n]);
    h[n] = 0.0f;
  }
  const float Dv = Ds[k*DIx + d];
  const float* dlt = delta + ((size_t)(b*KK + k)*DIx + d)*LL;
  const _Float16* ub = xch + (size_t)b*DIx*LL + (size_t)d*LL;
  const float* bcb = xdbl + (size_t)(b*KK + k)*LL*CDBL;
  float* yo = yb + ((size_t)(b*KK + k)*DIx + d)*LL;
  for (int l = 0; l < LL; ++l) {
    float dt = dlt[l];
    float u  = (float)ub[scan_perm(k, l)];
    const float* bc = bcb + (size_t)l*CDBL;
    float bu = dt * u, y = 0.0f;
#pragma unroll
    for (int n = 0; n < NS; ++n) {
      h[n] = __expf(dt * a[n]) * h[n] + bu * bc[RR + n];
      y   += h[n] * bc[RR + NS + n];
    }
    yo[l] = y + u * Dv;
  }
}

// ---------------- K7: cross-merge + out_norm LN + SiLU(z) gate -> yg f16 ----------------
__global__ void k_merge(const float* __restrict__ yb,   // (B,K,DI,L)
                        const float* __restrict__ xz,   // z = xz[:,192:]
                        const float* __restrict__ onw, const float* __restrict__ onb,
                        _Float16* __restrict__ yg) {    // (ROWS, DI)
  int row = blockIdx.x * 4 + (threadIdx.x >> 5);
  if (row >= ROWS) return;
  int b = row / LL, l = row % LL;
  int p1 = (l % WW)*WW + (l / WW);
  int lane = lane_id();
  float v[6], s = 0.f, s2 = 0.f;
#pragma unroll
  for (int j = 0; j < 6; ++j) {
    int d = lane + 32*j;
    float y = yb[(((size_t)b*KK + 0)*DIx + d)*LL + l]
            + yb[(((size_t)b*KK + 2)*DIx + d)*LL + (LL-1-l)]
            + yb[(((size_t)b*KK + 1)*DIx + d)*LL + p1]
            + yb[(((size_t)b*KK + 3)*DIx + d)*LL + (LL-1-p1)];
    v[j] = y; s += y; s2 += y*y;
  }
  s = wave_sum(s); s2 = wave_sum(s2);
  float mu = s * (1.0f/DIx);
  float var = s2 * (1.0f/DIx) - mu*mu;
  float rstd = rsqrtf(var + 1e-5f);
#pragma unroll
  for (int j = 0; j < 6; ++j) {
    int d = lane + 32*j;
    float ln = (v[j]-mu)*rstd*onw[d] + onb[d];
    float z = xz[(size_t)row*(2*DIx) + DIx + d];
    float g = z / (1.0f + __expf(-z));
    yg[(size_t)row*DIx + d] = (_Float16)(ln * g);
  }
}

// ---------------- K8: out_proj GEMM [4608x192]x[192x96] + residual ----------------
__global__ void k_gemm_outproj(const _Float16* __restrict__ yg,
                               const float* __restrict__ w,    // (96,192)
                               const float* __restrict__ tx,   // residual
                               float* __restrict__ x2) {
  const int NT = CC/16;                                        // 6
  int tile = blockIdx.x * 4 + (threadIdx.x >> 5);
  if (tile >= (ROWS/16)*NT) return;
  int mt = tile / NT, nt = tile % NT;
  v8f acc = {};
#pragma unroll
  for (int ks = 0; ks < DIx; ks += 32) {
    v16h a = frag_a_rm(yg + (size_t)mt*16*DIx + ks, DIx);
    v16h b = frag_b_wt(w + (size_t)nt*16*DIx + ks, DIx);
    acc = __builtin_amdgcn_wmma_f32_16x16x32_f16(false, a, false, b, (short)0, acc, false, false);
  }
  int lane = lane_id(), n = nt*16 + (lane & 15), hf = lane >> 4;
#pragma unroll
  for (int r = 0; r < 8; ++r) {
    int m = mt*16 + r + 8*hf;
    x2[(size_t)m*CC + n] = acc[r] + tx[(size_t)m*CC + n];
  }
}

// ---------------- K9: LayerNorm over 96 (LN2) ----------------
__global__ void k_ln_row96(const float* __restrict__ in, const float* __restrict__ w,
                           const float* __restrict__ bias, _Float16* __restrict__ out) {
  int row = blockIdx.x * 4 + (threadIdx.x >> 5);
  if (row >= ROWS) return;
  int lane = lane_id();
  float v[3], s = 0.f, s2 = 0.f;
#pragma unroll
  for (int j = 0; j < 3; ++j) {
    int c = lane + 32*j;
    float t = in[(size_t)row*CC + c];
    v[j] = t; s += t; s2 += t*t;
  }
  s = wave_sum(s); s2 = wave_sum(s2);
  float mu = s * (1.0f/CC);
  float var = s2 * (1.0f/CC) - mu*mu;
  float rstd = rsqrtf(var + 1e-5f);
#pragma unroll
  for (int j = 0; j < 3; ++j) {
    int c = lane + 32*j;
    out[(size_t)row*CC + c] = (_Float16)((v[j]-mu)*rstd*w[c] + bias[c]);
  }
}

// ---------------- K10: fc1 GEMM + bias + exact GELU ----------------
__global__ void k_gemm_fc1(const _Float16* __restrict__ h2, const float* __restrict__ w,
                           const float* __restrict__ bias, _Float16* __restrict__ g1) {
  const int NT = CC/16;
  int tile = blockIdx.x * 4 + (threadIdx.x >> 5);
  if (tile >= (ROWS/16)*NT) return;
  int mt = tile / NT, nt = tile % NT;
  v8f acc = {};
#pragma unroll
  for (int ks = 0; ks < CC; ks += 32) {
    v16h a = frag_a_rm(h2 + (size_t)mt*16*CC + ks, CC);
    v16h b = frag_b_wt(w + (size_t)nt*16*CC + ks, CC);
    acc = __builtin_amdgcn_wmma_f32_16x16x32_f16(false, a, false, b, (short)0, acc, false, false);
  }
  int lane = lane_id(), n = nt*16 + (lane & 15), hf = lane >> 4;
  float bv = bias[n];
#pragma unroll
  for (int r = 0; r < 8; ++r) {
    int m = mt*16 + r + 8*hf;
    float v = acc[r] + bv;
    v = 0.5f * v * (1.0f + erff(v * 0.70710678f));              // exact GELU
    g1[(size_t)m*CC + n] = (_Float16)v;
  }
}

// ---------------- K11: fc2 GEMM + bias + residual ----------------
__global__ void k_gemm_fc2(const _Float16* __restrict__ g1, const float* __restrict__ w,
                           const float* __restrict__ bias, const float* __restrict__ x2,
                           float* __restrict__ x3) {
  const int NT = CC/16;
  int tile = blockIdx.x * 4 + (threadIdx.x >> 5);
  if (tile >= (ROWS/16)*NT) return;
  int mt = tile / NT, nt = tile % NT;
  v8f acc = {};
#pragma unroll
  for (int ks = 0; ks < CC; ks += 32) {
    v16h a = frag_a_rm(g1 + (size_t)mt*16*CC + ks, CC);
    v16h b = frag_b_wt(w + (size_t)nt*16*CC + ks, CC);
    acc = __builtin_amdgcn_wmma_f32_16x16x32_f16(false, a, false, b, (short)0, acc, false, false);
  }
  int lane = lane_id(), n = nt*16 + (lane & 15), hf = lane >> 4;
  float bv = bias[n];
#pragma unroll
  for (int r = 0; r < 8; ++r) {
    int m = mt*16 + r + 8*hf;
    x3[(size_t)m*CC + n] = acc[r] + bv + x2[(size_t)m*CC + n];
  }
}

// ---------------- K12: 1x1 projector + sigmoid -> (B,1,H,W) ----------------
__global__ void k_proj(const float* __restrict__ x3, const float* __restrict__ pw,
                       const float* __restrict__ pb, float* __restrict__ out) {
  int row = blockIdx.x * blockDim.x + threadIdx.x;
  if (row >= ROWS) return;
  float acc = pb[0];
#pragma unroll 8
  for (int c = 0; c < CC; ++c) acc += x3[(size_t)row*CC + c] * pw[c];
  out[row] = 1.0f / (1.0f + __expf(-acc));
}

// ---------------- host-side launch ----------------
extern "C" void kernel_launch(void* const* d_in, const int* in_sizes, int n_in,
                              void* d_out, int out_size, void* d_ws, size_t ws_size,
                              hipStream_t stream) {
  const float* x          = (const float*)d_in[0];
  const float* ln1_w      = (const float*)d_in[1];
  const float* ln1_b      = (const float*)d_in[2];
  const float* in_proj_w  = (const float*)d_in[3];
  const float* conv_w     = (const float*)d_in[4];
  const float* conv_b     = (const float*)d_in[5];
  const float* x_proj_w   = (const float*)d_in[6];
  const float* dt_proj_w  = (const float*)d_in[7];
  const float* dt_proj_b  = (const float*)d_in[8];
  const float* A_log      = (const float*)d_in[9];
  const float* Ds         = (const float*)d_in[10];
  const float* out_norm_w = (const float*)d_in[11];
  const float* out_norm_b = (const float*)d_in[12];
  const float* out_proj_w = (const float*)d_in[13];
  const float* ln2_w      = (const float*)d_in[14];
  const float* ln2_b      = (const float*)d_in[15];
  const float* fc1_w      = (const float*)d_in[16];
  const float* fc1_b      = (const float*)d_in[17];
  const float* fc2_w      = (const float*)d_in[18];
  const float* fc2_b      = (const float*)d_in[19];
  const float* proj_w     = (const float*)d_in[20];
  const float* proj_b     = (const float*)d_in[21];
  float* out = (float*)d_out;

  // workspace carve-up (256B aligned)
  char* ws = (char*)d_ws;
  size_t off = 0;
  auto carve = [&](size_t bytes) -> char* {
    char* p = ws + off;
    off = (off + bytes + 255) & ~(size_t)255;
    return p;
  };
  float*    tx    = (float*)   carve((size_t)ROWS*CC*4);          // transposed input (residual)
  _Float16* h1h   = (_Float16*)carve((size_t)ROWS*CC*2);          // LN1 out, f16
  float*    xz    = (float*)   carve((size_t)ROWS*2*DIx*4);       // in_proj out
  _Float16* xch   = (_Float16*)carve((size_t)BB*DIx*LL*2);        // conv+SiLU, f16
  float*    xdbl  = (float*)   carve((size_t)BB*KK*LL*CDBL*4);    // x_proj out
  float*    delta = (float*)   carve((size_t)BB*KK*DIx*LL*4);     // softplus(dt)
  float*    ybuf  = (float*)   carve((size_t)BB*KK*DIx*LL*4);     // scan out
  _Float16* ygh   = (_Float16*)carve((size_t)ROWS*DIx*2);         // gated, f16
  float*    x2    = (float*)   carve((size_t)ROWS*CC*4);          // after SS2D residual
  _Float16* h2h   = (_Float16*)carve((size_t)ROWS*CC*2);          // LN2 out, f16
  _Float16* g1h   = (_Float16*)carve((size_t)ROWS*CC*2);          // gelu(fc1), f16
  float*    x3    = (float*)   carve((size_t)ROWS*CC*4);          // after MLP residual
  (void)ws_size; (void)in_sizes; (void)n_in; (void)out_size;

  // 1. transpose + LN1
  k_ln1<<<ROWS/4, 128, 0, stream>>>(x, ln1_w, ln1_b, tx, h1h);
  // 2. in_proj GEMM: 288 M-tiles x 24 N-tiles = 6912 tiles, 4 waves/block
  k_gemm_inproj<<<(ROWS/16)*((2*DIx)/16)/4, 128, 0, stream>>>(h1h, in_proj_w, xz);
  // 3. depthwise conv + SiLU
  k_conv<<<(BB*DIx*LL + 255)/256, 256, 0, stream>>>(xz, conv_w, conv_b, xch);
  // 4. x_proj GEMM (4 directions via index permutation): 3456 tiles
  k_gemm_xproj<<<(BB*KK*(LL/16)*3)/4, 128, 0, stream>>>(xch, x_proj_w, xdbl);
  // 5. dt_proj GEMM + softplus: 13824 tiles
  k_gemm_dtproj<<<(BB*KK*(LL/16)*(DIx/16))/4, 128, 0, stream>>>(xdbl, dt_proj_w, dt_proj_b, delta);
  // 6. selective scan: one thread per (b,k,d)
  k_scan<<<(BB*KK*DIx)/256, 256, 0, stream>>>(delta, xch, xdbl, A_log, Ds, ybuf);
  // 7. cross-merge + out_norm + gate
  k_merge<<<ROWS/4, 128, 0, stream>>>(ybuf, xz, out_norm_w, out_norm_b, ygh);
  // 8. out_proj GEMM + residual: 1728 tiles
  k_gemm_outproj<<<(ROWS/16)*(CC/16)/4, 128, 0, stream>>>(ygh, out_proj_w, tx, x2);
  // 9. LN2
  k_ln_row96<<<ROWS/4, 128, 0, stream>>>(x2, ln2_w, ln2_b, h2h);
  // 10. fc1 + GELU
  k_gemm_fc1<<<(ROWS/16)*(CC/16)/4, 128, 0, stream>>>(h2h, fc1_w, fc1_b, g1h);
  // 11. fc2 + residual
  k_gemm_fc2<<<(ROWS/16)*(CC/16)/4, 128, 0, stream>>>(g1h, fc2_w, fc2_b, x2, x3);
  // 12. 1x1 projector + sigmoid
  k_proj<<<(ROWS + 255)/256, 256, 0, stream>>>(x3, proj_w, proj_b, out);
}
